// WindowBasedGraphAttentionBlock_627065225606
// MI455X (gfx1250) — matrix operand order repaired and checked by reading, not compiled, not run on hardware
//
#include <hip/hip_runtime.h>
#include <cstddef>

// ---------------------------------------------------------------------------
// Problem constants (exact, from the reference): all tile sizes divide evenly.
// ---------------------------------------------------------------------------
#define NTOK   41472      // tokens
#define CDIM   512
#define NWIN   162
#define WSZ    256        // window size (NTOK / NWIN)
#define NH     8
#define HD     64
#define HID    2048       // FFN hidden

static const size_t QKV_ELEMS = (size_t)NWIN * NH * WSZ * HD;   // 21,233,664

typedef __attribute__((ext_vector_type(8)))  float  v8f;
typedef __attribute__((ext_vector_type(16))) __bf16 v16bf;

union Frag16 { v16bf v; unsigned u[8]; };

__device__ __forceinline__ unsigned short f2bf(float f) {
  unsigned u = __float_as_uint(f);
  unsigned r = u + 0x7FFFu + ((u >> 16) & 1u);   // round-to-nearest-even
  return (unsigned short)(r >> 16);
}

__device__ __forceinline__ v8f wmma_bf16(const Frag16& a, const Frag16& b, v8f c) {
  return __builtin_amdgcn_wmma_f32_16x16x32_bf16(false, a.v, false, b.v,
                                                 (short)0, c, false, false);
}

// CDNA5 async global->LDS copy (16B per lane), tracked by ASYNCcnt.
// The "memory" clobber orders it against the later ds reads; the visible
// volatile LDS stores in each kernel keep the optimizer from treating the
// async-written tiles as unwritten memory.
__device__ __forceinline__ void async_copy_b128(void* lds_ptr, const void* gaddr) {
  asm volatile("global_load_async_to_lds_b128 %0, %1, off"
               :: "v"((unsigned)(unsigned long long)lds_ptr), "v"(gaddr)
               : "memory");
}
__device__ __forceinline__ void async_wait0()   { asm volatile("s_wait_asynccnt 0x0" ::: "memory"); }
__device__ __forceinline__ void async_wait_le4(){ asm volatile("s_wait_asynccnt 0x4" ::: "memory"); }
__device__ __forceinline__ void ds_wait0()      { asm volatile("s_wait_dscnt 0x0"    ::: "memory"); }

// ---------------------------------------------------------------------------
// 0) fp32 -> bf16 with k-pair packing: out DWORD r*N+n = {W[2r][n], W[2r+1][n]}
//    This is exactly the CDNA5 16-bit B-fragment register layout, so GEMM
//    B tiles become straight async copies.
// ---------------------------------------------------------------------------
__global__ void cvt_pack_kernel(const float* __restrict__ src,
                                unsigned* __restrict__ dst, int K2, int N) {
  int i = blockIdx.x * 256 + threadIdx.x;
  if (i < K2 * N) {
    int r = i / N, n = i - r * N;
    unsigned lo = f2bf(src[(size_t)(2 * r)     * N + n]);
    unsigned hi = f2bf(src[(size_t)(2 * r + 1) * N + n]);
    dst[i] = lo | (hi << 16);
  }
}

// ---------------------------------------------------------------------------
// 1) Deterministic window grouping: one wave (32 lanes) per window scans all
//    tokens in order; ballot + popc gives a stable counting sort.
// ---------------------------------------------------------------------------
__global__ void perm_kernel(const int* __restrict__ wids, int* __restrict__ perm) {
  int w = blockIdx.x;
  int lane = threadIdx.x;            // 32 threads
  int count = 0;
  for (int base = 0; base < NTOK; base += 32) {
    int n = base + lane;
    bool m = (wids[n] == w);
    unsigned mask = (unsigned)__ballot(m);
    if (m) perm[w * WSZ + count + __popc(mask & ((1u << lane) - 1u))] = n;
    count += __popc(mask);
  }
}

// ---------------------------------------------------------------------------
// 2) LayerNorm -> bf16.  One block (256 threads) per output row.
// ---------------------------------------------------------------------------
__global__ __launch_bounds__(256)
void ln_kernel(const float* __restrict__ x, const float* __restrict__ g,
               const float* __restrict__ b, const int* __restrict__ perm,
               unsigned short* __restrict__ outB) {
  int row = blockIdx.x;
  int token = perm ? perm[row] : row;
  const float* xr = x + (size_t)token * CDIM;
  int tid = threadIdx.x;
  float v1 = xr[tid], v2 = xr[tid + 256];
  float s = v1 + v2, q = v1 * v1 + v2 * v2;
  #pragma unroll
  for (int d = 16; d; d >>= 1) { s += __shfl_xor(s, d, 32); q += __shfl_xor(q, d, 32); }
  __shared__ float ps[8], pq[8];
  int wv = tid >> 5, lane = tid & 31;
  if (lane == 0) { ps[wv] = s; pq[wv] = q; }
  __syncthreads();
  s = 0.f; q = 0.f;
  #pragma unroll
  for (int i = 0; i < 8; ++i) { s += ps[i]; q += pq[i]; }
  float mean = s * (1.0f / CDIM);
  float var  = q * (1.0f / CDIM) - mean * mean;
  float inv  = rsqrtf(var + 1e-5f);
  outB[(size_t)row * CDIM + tid]       = f2bf((v1 - mean) * inv * g[tid]       + b[tid]);
  outB[(size_t)row * CDIM + tid + 256] = f2bf((v2 - mean) * inv * g[tid + 256] + b[tid + 256]);
}

// ---------------------------------------------------------------------------
// 3) Generic bf16 WMMA GEMM, 128x128x32 block tile, 8 waves (wave = 64x32).
//    Double-buffered async staging (GLOBAL_LOAD_ASYNC_TO_LDS_B128): tile k+1
//    is in flight while tile k feeds the WMMAs; s_wait_asynccnt 4 separates.
// ---------------------------------------------------------------------------
#define MODE_QKV  0
#define MODE_PROJ 1
#define MODE_GELU 2
#define MODE_FFN2 3

template <int MODE, int KDIM, int NDIM>
__global__ __launch_bounds__(256)
void gemm_kernel(const unsigned short* __restrict__ A,     // [M, KDIM] bf16
                 const unsigned* __restrict__ Bp,          // [KDIM/2, NDIM] k-pair DWORDs
                 const float* __restrict__ bias,           // [NDIM]
                 const float* __restrict__ resid,          // PROJ: x ; FFN2: y
                 const int* __restrict__ perm,             // PROJ only
                 unsigned short* __restrict__ outB,        // QKV / GELU
                 float* __restrict__ outF) {               // PROJ / FFN2
  __shared__ unsigned Asm[2][128 * 36];  // 128 rows x 32 bf16, stride 144B (16B align)
  __shared__ unsigned Bsm[2][16 * 132];  // 16 k-pairs x 128 cols + pad, stride 528B

  const int tid  = threadIdx.x;
  const int lane = tid & 31, wid = tid >> 5;
  const int wm = wid >> 2, wn = wid & 3;     // 2 x 4 wave grid
  const int hb = lane >> 4, l16 = lane & 15;
  const int m0 = blockIdx.y * 128, n0 = blockIdx.x * 128;

  // Visible LDS writes: the async-written tiles must not look "never stored"
  // to the optimizer (prevents undef-folding of the fragment ds loads).
  *(volatile unsigned*)&Asm[0][tid] = 0u;
  *(volatile unsigned*)&Asm[1][tid] = 0u;
  *(volatile unsigned*)&Bsm[0][tid] = 0u;
  *(volatile unsigned*)&Bsm[1][tid] = 0u;
  ds_wait0();   // zeros land before the async pipe can write the same lines

  // Per-thread async-copy addresses
  const int ar = tid >> 1, ah = tid & 1;          // A: row, 16-bf16 half
  const int bkp = tid >> 4, bc0 = (tid & 15) * 8; // B: k-pair row, 8-DWORD chunk
  const char* a_g = (const char*)(A + (size_t)(m0 + ar) * KDIM + ah * 16);
  const char* b_g = (const char*)(Bp + (size_t)bkp * NDIM + n0 + bc0);

  auto stage = [&](int buf, int ks) {
    const char* ag = a_g + (size_t)ks * 64;                   // 32 bf16 per step
    const char* bg = b_g + (size_t)ks * (16 * NDIM * 4);      // 16 k-pair rows
    async_copy_b128(&Asm[buf][ar * 36 + ah * 8],     ag);
    async_copy_b128(&Asm[buf][ar * 36 + ah * 8 + 4], ag + 16);
    async_copy_b128(&Bsm[buf][bkp * 132 + bc0],      bg);
    async_copy_b128(&Bsm[buf][bkp * 132 + bc0 + 4],  bg + 16);
  };

  v8f acc[4][2];
  #pragma unroll
  for (int mt = 0; mt < 4; ++mt)
    #pragma unroll
    for (int nt = 0; nt < 2; ++nt)
      acc[mt][nt] = (v8f){0.f, 0.f, 0.f, 0.f, 0.f, 0.f, 0.f, 0.f};

  constexpr int KSTEPS = KDIM / 32;
  stage(0, 0);
  #pragma unroll 4
  for (int ks = 0; ks < KSTEPS; ++ks) {
    const int cur = ks & 1;
    if (ks + 1 < KSTEPS) { stage(cur ^ 1, ks + 1); async_wait_le4(); }
    else                 { async_wait0(); }
    __syncthreads();

    Frag16 af[4], bfr[2];
    #pragma unroll
    for (int mt = 0; mt < 4; ++mt) {
      int mrow = wm * 64 + mt * 16 + l16;
      #pragma unroll
      for (int v = 0; v < 8; ++v) {
        int kb = hb * 16 + v * 2;   // contiguous k order (matches B)
        af[mt].u[v] = Asm[cur][mrow * 36 + (kb >> 1)];
      }
    }
    #pragma unroll
    for (int nt = 0; nt < 2; ++nt) {
      int ncol = wn * 32 + nt * 16 + l16;
      #pragma unroll
      for (int v = 0; v < 8; ++v)
        bfr[nt].u[v] = Bsm[cur][(hb * 8 + v) * 132 + ncol]; // contiguous k order (matches A)
    }
    #pragma unroll
    for (int mt = 0; mt < 4; ++mt)
      #pragma unroll
      for (int nt = 0; nt < 2; ++nt)
        acc[mt][nt] = wmma_bf16(af[mt], bfr[nt], acc[mt][nt]);
    ds_wait0();          // all our ds reads landed before anyone overwrites LDS
    __syncthreads();
  }

  // ---------------- fused epilogues -----------------
  #pragma unroll
  for (int mt = 0; mt < 4; ++mt) {
    #pragma unroll
    for (int nt = 0; nt < 2; ++nt) {
      int col = n0 + wn * 32 + nt * 16 + l16;
      float bv = bias[col];
      #pragma unroll
      for (int v = 0; v < 8; ++v) {
        int row = m0 + wm * 64 + mt * 16 + hb * 8 + v;   // C-layout row
        float val = acc[mt][nt][v] + bv;
        if constexpr (MODE == MODE_QKV) {
          int which = col >> 9, rem = col & 511;
          int hh = rem >> 6, d = rem & 63;
          int w = row >> 8, wr = row & 255;
          if (which == 2) {
            // V stored k-pair packed: DWORD (wr/2)*HD+d = {row 2r, row 2r+1}
            outB[2 * QKV_ELEMS + (size_t)(w * NH + hh) * (WSZ * HD) +
                 (size_t)(wr >> 1) * (2 * HD) + d * 2 + (wr & 1)] = f2bf(val);
          } else {
            outB[(size_t)which * QKV_ELEMS +
                 (((size_t)(w * NH + hh) * WSZ + wr) * HD) + d] = f2bf(val);
          }
        } else if constexpr (MODE == MODE_PROJ) {
          int token = perm[row];
          size_t idx = (size_t)token * CDIM + col;
          outF[idx] = resid[idx] + val;
        } else if constexpr (MODE == MODE_GELU) {
          float gl = 0.5f * val * (1.0f + erff(val * 0.70710678118654752f));
          outB[(size_t)row * HID + col] = f2bf(gl);
        } else {  // MODE_FFN2
          size_t idx = (size_t)row * CDIM + col;
          outF[idx] = resid[idx] + val;
        }
      }
    }
  }
}

// ---------------------------------------------------------------------------
// 4) Windowed attention: one workgroup per (window, head); 8 waves.
//    K + (pre-packed) V staged by async copies; each wave owns 32 q rows
//    (two 16-row strips): S = QK^T via WMMA, softmax via intra-half shuffles,
//    P staged in per-wave LDS, O = P V via WMMA.
//    LDS: 36,864 (K) + 34,816 (Vp) + 8*8,448 (P) = 139,264 B (<320KB/WG).
// ---------------------------------------------------------------------------
__global__ __launch_bounds__(256)
void attn_kernel(const unsigned short* __restrict__ qkvB,
                 unsigned short* __restrict__ attnB,
                 const float* __restrict__ rel_bias) {
  __shared__ unsigned smem[34816];
  const int KLS = 0;            // 256 rows * 36 uints (stride 144B)
  const int VP  = 9216;         // 128 kpairs * 68 uints (stride 272B)
  const int PLS = 17920;        // 8 waves * (16 rows * 132 uints)

  const int tid = threadIdx.x, lane = tid & 31, wid = tid >> 5;
  const int hb = lane >> 4, l16 = lane & 15;
  const int w = blockIdx.x >> 3, h = blockIdx.x & 7;

  const unsigned* qU = (const unsigned*)qkvB + (size_t)(w * NH + h) * WSZ * 32;
  const char* kG = (const char*)qkvB + 2 * QKV_ELEMS + (size_t)(w * NH + h) * (WSZ * HD * 2);
  const char* vG = (const char*)qkvB + 4 * QKV_ELEMS + (size_t)(w * NH + h) * (WSZ * HD * 2);

  // Visible touch of the async-written regions (see GEMM comment).
  *(volatile unsigned*)&smem[KLS + tid] = 0u;
  *(volatile unsigned*)&smem[VP + tid]  = 0u;
  ds_wait0();

  {   // K: 256 rows x 128B, one row per thread -> 8 async b128
    const char* src = kG + (size_t)tid * 128;
    #pragma unroll
    for (int i = 0; i < 8; ++i)
      async_copy_b128((char*)&smem[KLS + tid * 36] + i * 16, src + i * 16);
  }
  {   // V (k-pair packed): 128 rows x 256B, half-row per thread -> 8 async b128
    int kp = tid >> 1, dp = tid & 1;
    const char* src = vG + (size_t)kp * 256 + dp * 128;
    #pragma unroll
    for (int i = 0; i < 8; ++i)
      async_copy_b128((char*)&smem[VP + kp * 68 + dp * 32] + i * 16, src + i * 16);
  }
  async_wait0();
  __syncthreads();

  const float bias_s = rel_bias[h];
  unsigned short* P16 = (unsigned short*)&smem[PLS + wid * 2112];

  #pragma unroll
  for (int s = 0; s < 2; ++s) {
    const int qr0 = wid * 32 + s * 16;

    // Q A-fragments (16 rows x 64), straight from global (K = 2 chunks of 32)
    Frag16 aq[2];
    #pragma unroll
    for (int kc = 0; kc < 2; ++kc)
      #pragma unroll
      for (int v = 0; v < 8; ++v) {
        int kb = kc * 32 + hb * 16 + v * 2;
        aq[kc].u[v] = qU[(size_t)(qr0 + l16) * 32 + (kb >> 1)];
      }

    // S strip: 16 x 256
    v8f sacc[16];
    #pragma unroll
    for (int nt = 0; nt < 16; ++nt) {
      sacc[nt] = (v8f){0.f, 0.f, 0.f, 0.f, 0.f, 0.f, 0.f, 0.f};
      #pragma unroll
      for (int kc = 0; kc < 2; ++kc) {
        Frag16 bk;
        #pragma unroll
        for (int v = 0; v < 8; ++v)
          bk.u[v] = smem[KLS + (nt * 16 + l16) * 36 + kc * 16 + hb * 8 + v];
        sacc[nt] = wmma_bf16(aq[kc], bk, sacc[nt]);
      }
    }

    // Softmax per row (row M = v + 8*hb lives on the 16 lanes of one half)
    #pragma unroll
    for (int v = 0; v < 8; ++v) {
      float m = -3.0e38f;
      #pragma unroll
      for (int t = 0; t < 16; ++t) {
        float xv = sacc[t][v] * 0.125f + bias_s;
        sacc[t][v] = xv;
        m = fmaxf(m, xv);
      }
      m = fmaxf(m, __shfl_xor(m, 1, 32));
      m = fmaxf(m, __shfl_xor(m, 2, 32));
      m = fmaxf(m, __shfl_xor(m, 4, 32));
      m = fmaxf(m, __shfl_xor(m, 8, 32));
      float ssum = 0.f;
      #pragma unroll
      for (int t = 0; t < 16; ++t) {
        float e = __expf(sacc[t][v] - m);
        sacc[t][v] = e;
        ssum += e;
      }
      ssum += __shfl_xor(ssum, 1, 32);
      ssum += __shfl_xor(ssum, 2, 32);
      ssum += __shfl_xor(ssum, 4, 32);
      ssum += __shfl_xor(ssum, 8, 32);
      float inv = 1.0f / ssum;
      int M = v + hb * 8;
      #pragma unroll
      for (int t = 0; t < 16; ++t)
        P16[M * 264 + t * 16 + l16] = f2bf(sacc[t][v] * inv);
    }
    __syncthreads();

    // O strip = P (16x256) x V (256x64)
    #pragma unroll
    for (int dt = 0; dt < 4; ++dt) {
      v8f oacc = (v8f){0.f, 0.f, 0.f, 0.f, 0.f, 0.f, 0.f, 0.f};
      #pragma unroll
      for (int kc = 0; kc < 8; ++kc) {
        Frag16 ap, bv;
        #pragma unroll
        for (int v = 0; v < 8; ++v) {
          int kb = hb * 16 + v * 2;
          ap.u[v] = smem[PLS + wid * 2112 + l16 * 132 + kc * 16 + (kb >> 1)];
          bv.u[v] = smem[VP + (kc * 16 + hb * 8 + v) * 68 + dt * 16 + l16];
        }
        oacc = wmma_bf16(ap, bv, oacc);
      }
      #pragma unroll
      for (int v = 0; v < 8; ++v) {
        int row = qr0 + v + hb * 8;
        int col = h * HD + dt * 16 + l16;
        attnB[(size_t)(w * WSZ + row) * CDIM + col] = f2bf(oacc[v]);
      }
    }
    __syncthreads();
  }
}

// ---------------------------------------------------------------------------
// Host orchestration
// ---------------------------------------------------------------------------
extern "C" void kernel_launch(void* const* d_in, const int* in_sizes, int n_in,
                              void* d_out, int out_size, void* d_ws, size_t ws_size,
                              hipStream_t stream) {
  (void)in_sizes; (void)n_in; (void)out_size; (void)ws_size;
  const float* x       = (const float*)d_in[0];
  const float* mesh    = (const float*)d_in[1];
  const float* ln1_g   = (const float*)d_in[2];
  const float* ln1_b   = (const float*)d_in[3];
  const float* qkv_w   = (const float*)d_in[4];
  const float* qkv_b   = (const float*)d_in[5];
  const float* rel_b   = (const float*)d_in[6];
  const float* proj_w  = (const float*)d_in[7];
  const float* proj_b  = (const float*)d_in[8];
  const float* ln2_g   = (const float*)d_in[9];
  const float* ln2_b   = (const float*)d_in[10];
  const float* ffn_w1  = (const float*)d_in[11];
  const float* ffn_b1  = (const float*)d_in[12];
  const float* ffn_w2  = (const float*)d_in[13];
  const float* ffn_b2  = (const float*)d_in[14];
  const int*   wids    = (const int*)d_in[15];
  float* outF = (float*)d_out;

  // Workspace layout (bytes, all 16B-aligned)
  char* ws = (char*)d_ws;
  const size_t OFF_PERM = 0;                                     //   165,888
  const size_t OFF_H    = 165888;                                //  42,467,328 (hB, later attnB)
  const size_t OFF_QKV  = OFF_H + (size_t)NTOK * CDIM * 2;       // 127,401,984 (qkv, later h2B)
  const size_t OFF_Y    = OFF_QKV + 3 * QKV_ELEMS * 2;           //  84,934,656
  const size_t OFF_F1   = OFF_Y + (size_t)NTOK * CDIM * 4;       // 169,869,312
  const size_t OFF_WB   = OFF_F1 + (size_t)NTOK * HID * 2;       // packed bf16 weights

  int*            perm   = (int*)(ws + OFF_PERM);
  unsigned short* hB     = (unsigned short*)(ws + OFF_H);
  unsigned short* attnB  = hB;                                   // reuse after QKV GEMM
  unsigned short* qkvB   = (unsigned short*)(ws + OFF_QKV);
  unsigned short* h2B    = qkvB;                                 // reuse after attention
  float*          y      = (float*)(ws + OFF_Y);
  unsigned short* f1B    = (unsigned short*)(ws + OFF_F1);
  unsigned*       qkv_wP = (unsigned*)(ws + OFF_WB);             // [K/2, N] DWORDs each
  unsigned*       proj_wP= qkv_wP + (size_t)(CDIM / 2) * (3 * CDIM);
  unsigned*       ffn1_wP= proj_wP + (size_t)(CDIM / 2) * CDIM;
  unsigned*       ffn2_wP= ffn1_wP + (size_t)(CDIM / 2) * HID;

  // Weight conversion fp32 -> k-pair-packed bf16
  {
    int n1 = (CDIM / 2) * 3 * CDIM, n2 = (CDIM / 2) * CDIM;
    int n3 = (CDIM / 2) * HID,      n4 = (HID / 2) * CDIM;
    cvt_pack_kernel<<<(n1 + 255) / 256, 256, 0, stream>>>(qkv_w,  qkv_wP, CDIM / 2, 3 * CDIM);
    cvt_pack_kernel<<<(n2 + 255) / 256, 256, 0, stream>>>(proj_w, proj_wP, CDIM / 2, CDIM);
    cvt_pack_kernel<<<(n3 + 255) / 256, 256, 0, stream>>>(ffn_w1, ffn1_wP, CDIM / 2, HID);
    cvt_pack_kernel<<<(n4 + 255) / 256, 256, 0, stream>>>(ffn_w2, ffn2_wP, HID / 2, CDIM);
  }

  // 1) window grouping
  perm_kernel<<<NWIN, 32, 0, stream>>>(wids, perm);

  // 2) LN1 (gathered rows) -> hB
  ln_kernel<<<NTOK, 256, 0, stream>>>(x, ln1_g, ln1_b, perm, hB);

  // 3) QKV GEMM: [NTOK,512] x [512,1536] -> per-(window,head) q/k/v (v packed)
  gemm_kernel<MODE_QKV, CDIM, 3 * CDIM><<<dim3(3 * CDIM / 128, NTOK / 128), 256, 0, stream>>>(
      hB, qkv_wP, qkv_b, nullptr, nullptr, qkvB, nullptr);

  // 4) attention per (window, head)
  attn_kernel<<<NWIN * NH, 256, 0, stream>>>(qkvB, attnB, rel_b);

  // 5) proj GEMM + residual scatter -> y (original token order)
  gemm_kernel<MODE_PROJ, CDIM, CDIM><<<dim3(CDIM / 128, NTOK / 128), 256, 0, stream>>>(
      attnB, proj_wP, proj_b, x, perm, nullptr, y);

  // 6) LN2 -> h2B
  ln_kernel<<<NTOK, 256, 0, stream>>>(y, ln2_g, ln2_b, nullptr, h2B);

  // 7) FFN1 + exact GELU -> f1B
  gemm_kernel<MODE_GELU, CDIM, HID><<<dim3(HID / 128, NTOK / 128), 256, 0, stream>>>(
      h2B, ffn1_wP, ffn_b1, nullptr, nullptr, f1B, nullptr);

  // 8) FFN2 + residual -> d_out
  gemm_kernel<MODE_FFN2, HID, CDIM><<<dim3(CDIM / 128, NTOK / 128), 256, 0, stream>>>(
      f1B, ffn2_wP, ffn_b2, y, nullptr, nullptr, outF);

  // 9) mesh passthrough (second tuple output)
  hipMemcpyAsync(outF + (size_t)NTOK * CDIM, mesh, (size_t)NTOK * 3 * sizeof(float),
                 hipMemcpyDeviceToDevice, stream);
}
